// UpSampleInterpolation_21844203667629
// MI455X (gfx1250) — matrix operand order repaired and checked by reading, not compiled, not run on hardware
//
#include <hip/hip_runtime.h>
#include <hip/hip_bf16.h>

#define BB 8
#define CC 128
#define NN 8192
#define MM 2048

typedef __attribute__((ext_vector_type(8)))  __bf16 v8bf;
typedef __attribute__((ext_vector_type(16))) __bf16 v16bf;
typedef __attribute__((ext_vector_type(8)))  float  v8f;

#define DEV static __device__ __forceinline__

__host__ __device__ constexpr int ilog2c(int v) { int r = 0; while (v >>= 1) r++; return r; }

DEV v8f wmma_bf16(v16bf a, v16bf b, v8f c) {
  return __builtin_amdgcn_wmma_f32_16x16x32_bf16(false, a, false, b, (short)0, c, false, false);
}

DEV v8f wmma3(v16bf ah, v16bf al, v16bf bh, v16bf bl, v8f c) {
  c = wmma_bf16(ah, bh, c);
  c = wmma_bf16(al, bh, c);
  c = wmma_bf16(ah, bl, c);
  return c;
}

// A-matrix (16x32, MxK) lane load. Row-major source [row][k], rowStride elems.
// Lane half h: K in {h*16..h*16+15} -> two contiguous 16B chunks.
DEV v16bf load_Amat(const __bf16* __restrict__ base, int rowStride, int r0, int k0, int lane) {
  int row = lane & 15, h = lane >> 4;
  const __bf16* p = base + (size_t)(r0 + row) * rowStride + k0 + h * 16;
  v8bf a0 = *(const v8bf*)(p);
  v8bf a1 = *(const v8bf*)(p + 8);
  v16bf a;
#pragma unroll
  for (int i = 0; i < 8; i++) { a[i] = a0[i]; a[8 + i] = a1[i]; }
  return a;
}

DEV void split_bf16(float x, __bf16* hi, __bf16* lo) {
  __bf16 h = (__bf16)x;
  *hi = h;
  *lo = (__bf16)(x - (float)h);
}

// ---------------- K0: weights -> bf16 hi/lo ----------------
__global__ void k_prep_weights(const float* __restrict__ wc, const float* __restrict__ wr,
                               __bf16* wc_hi, __bf16* wc_lo, __bf16* wr_hi, __bf16* wr_lo) {
  int i = blockIdx.x * 256 + threadIdx.x;
  const int nc = CC * CC;
  const int nr = CC * 2 * CC;
  if (i < nc) split_bf16(wc[i], wc_hi + i, wc_lo + i);
  else if (i < nc + nr) { int j = i - nc; split_bf16(wr[j], wr_hi + j, wr_lo + j); }
}

// ---------------- K1: LDS-tiled transpose f32 [b][Cc][Lc] -> bf16 hi/lo [b][Lc][RS] ----
template <int Cc, int Lc, int RS>
__global__ void k_transpose_split(const float* __restrict__ src, __bf16* __restrict__ hi,
                                  __bf16* __restrict__ lo) {
  __shared__ float t[32][33];
  constexpr int tilesL = Lc >> 5;
  constexpr int tilesC = Cc >> 5;
  int bid = blockIdx.x;
  int tl = bid & (tilesL - 1); bid >>= ilog2c(tilesL);
  int tc = bid & (tilesC - 1); int b = bid >> ilog2c(tilesC);
  int l0 = tl * 32, c0 = tc * 32;
  int tx = threadIdx.x & 31, ty = threadIdx.x >> 5;  // 32x8
#pragma unroll
  for (int j = 0; j < 4; j++) {
    int c = c0 + ty + j * 8;
    t[ty + j * 8][tx] = src[((size_t)b * Cc + c) * Lc + l0 + tx];
  }
  __syncthreads();
#pragma unroll
  for (int j = 0; j < 4; j++) {
    int l = l0 + ty + j * 8;
    float v = t[tx][ty + j * 8];
    size_t o = ((size_t)b * Lc + l) * RS + c0 + tx;
    __bf16 h = (__bf16)v;
    hi[o] = h;
    lo[o] = (__bf16)(v - (float)h);
  }
}

// ---------------- K2: per-column squared norms: dst[b][l] = sum_c src[b][c][l]^2 ------
template <int Cc, int Lc>
__global__ void k_colnorm2(const float* __restrict__ src, float* __restrict__ dst) {
  int i = blockIdx.x * 256 + threadIdx.x;
  int b = i >> ilog2c(Lc), l = i & (Lc - 1);
  const float* p = src + (size_t)b * Cc * Lc + l;
  float s = 0.f;
#pragma unroll 8
  for (int c = 0; c < Cc; c++) { float v = p[(size_t)c * Lc]; s += v * v; }
  dst[i] = s;
}

// ---------------- K3: bf16x3 WMMA conv. Block = 8 waves = all 8 o-tiles of one l-tile;
// the B panel [16 x Kc] (hi/lo) is staged once in LDS and shared by all waves. --------
template <int Ob, int Lc, int Kc, int RS>
__global__ void k_conv_wmma(const __bf16* __restrict__ a_hi, const __bf16* __restrict__ a_lo,
                            const __bf16* __restrict__ bt_hi, const __bf16* __restrict__ bt_lo,
                            float* __restrict__ out) {
  constexpr int tilesL = Lc / 16;
  __shared__ __align__(32) __bf16 Bh_s[16 * Kc];
  __shared__ __align__(32) __bf16 Bl_s[16 * Kc];
  int lane = threadIdx.x & 31;
  int w = threadIdx.x >> 5;  // o-tile 0..7 (Ob == 128)
  int bid = blockIdx.x;
  int tl = bid & (tilesL - 1);
  int b = bid >> ilog2c(tilesL);
  int o0 = w * 16, l0 = tl * 16;

  const __bf16* bh_g = bt_hi + ((size_t)b * Lc + l0) * RS;
  const __bf16* bl_g = bt_lo + ((size_t)b * Lc + l0) * RS;
  constexpr int chunks = 16 * Kc / 8;  // 16B v8bf chunks per panel
  for (int e = threadIdx.x; e < chunks; e += 256) {
    int row = e / (Kc / 8);
    int off = (e % (Kc / 8)) * 8;
    *(v8bf*)(Bh_s + row * Kc + off) = *(const v8bf*)(bh_g + (size_t)row * RS + off);
    *(v8bf*)(Bl_s + row * Kc + off) = *(const v8bf*)(bl_g + (size_t)row * RS + off);
  }
  __syncthreads();

  int col = lane & 15, h = lane >> 4;
  v8f acc = {};
#pragma unroll
  for (int k0 = 0; k0 < Kc; k0 += 32) {
    v16bf ah = load_Amat(a_hi, Kc, o0, k0, lane);
    v16bf al = load_Amat(a_lo, Kc, o0, k0, lane);
    v16bf bh = *(const v16bf*)(Bh_s + col * Kc + k0 + h * 16);
    v16bf bl = *(const v16bf*)(Bl_s + col * Kc + k0 + h * 16);
    acc = wmma3(ah, al, bh, bl, acc);
  }
#pragma unroll
  for (int r = 0; r < 8; r++)
    out[((size_t)b * Ob + o0 + r + h * 8) * Lc + l0 + col] = acc[r];
}

// ---------------- K4: BN stats (training-mode, biased var) -> scale/shift -------------
template <int Lc>
__global__ void k_bn_stats(const float* __restrict__ src, const float* __restrict__ g,
                           const float* __restrict__ beta, float* __restrict__ scale,
                           float* __restrict__ shift) {
  __shared__ float ssum[256], ssq[256];
  int ch = blockIdx.x;
  float s = 0.f, q = 0.f;
  for (int b = 0; b < BB; b++) {
    const float* p = src + ((size_t)b * CC + ch) * Lc;
    for (int l = threadIdx.x; l < Lc; l += 256) { float v = p[l]; s += v; q += v * v; }
  }
  ssum[threadIdx.x] = s; ssq[threadIdx.x] = q;
  __syncthreads();
  for (int st = 128; st; st >>= 1) {
    if (threadIdx.x < st) { ssum[threadIdx.x] += ssum[threadIdx.x + st]; ssq[threadIdx.x] += ssq[threadIdx.x + st]; }
    __syncthreads();
  }
  if (threadIdx.x == 0) {
    constexpr float cnt = (float)BB * (float)Lc;
    float mu = ssum[0] / cnt;
    float var = ssq[0] / cnt - mu * mu;
    float sc = g[ch] * rsqrtf(var + 1e-5f);
    scale[ch] = sc;
    shift[ch] = beta[ch] - mu * sc;
  }
}

// ---------------- K5: apply BN+LeakyReLU to psc and transpose -> psc_t [b][m][c] f32 --
__global__ void k_bn_apply_psct(const float* __restrict__ pre, const float* __restrict__ scale,
                                const float* __restrict__ shift, float* __restrict__ psc_t) {
  int i = blockIdx.x * 256 + threadIdx.x;  // (b*MM + m)*CC + c
  int c = i & (CC - 1);
  int bm = i >> 7;
  int m = bm & (MM - 1);
  int b = bm >> 11;
  float y = pre[((size_t)b * CC + c) * MM + m] * scale[c] + shift[c];
  psc_t[i] = y > 0.f ? y : 0.2f * y;
}

DEV void top3_insert(float d, int gi, float& d0, int& i0, float& d1, int& i1, float& d2, int& i2) {
  if (!((d < d2) || (d == d2 && gi < i2))) return;
  bool lt0 = (d < d0) || (d == d0 && gi < i0);
  bool lt1 = (d < d1) || (d == d1 && gi < i1);
  if (lt0)      { d2 = d1; i2 = i1; d1 = d0; i1 = i0; d0 = d; i0 = gi; }
  else if (lt1) { d2 = d1; i2 = i1; d1 = d;  i1 = gi; }
  else          { d2 = d;  i2 = gi; }
}

// ---------------- K6: fused distance GEMM + top-3. Block = 8 waves (8 n-tiles, same b);
// each 16x128 B panel (hi/lo) staged once per block in LDS, prefetch next panel. -------
__global__ void k_dist_topk(const __bf16* __restrict__ up_hi, const __bf16* __restrict__ up_lo,
                            const __bf16* __restrict__ ps_hi, const __bf16* __restrict__ ps_lo,
                            const float* __restrict__ q2, const float* __restrict__ k2,
                            int* __restrict__ idx_out, float* __restrict__ w_out) {
  __shared__ __align__(32) __bf16 Bh_s[16 * CC];
  __shared__ __align__(32) __bf16 Bl_s[16 * CC];
  __shared__ float tile[8][16][17];
  __shared__ float md[8][32][3];
  __shared__ int   mi[8][32][3];
  int lane = threadIdx.x & 31;
  int w = threadIdx.x >> 5;
  int wg = blockIdx.x * 8 + w;
  int tn = wg & (NN / 16 - 1);   // 512 tiles per batch
  int b = wg >> ilog2c(NN / 16);
  int n0 = tn * 16;
  int col = lane & 15, h = lane >> 4;
  int cbase = h * 8;  // column half this lane scans; scan row = col

  const __bf16* uh = up_hi + (size_t)b * NN * (2 * CC);  // x_t rows, stride 256, cols 0..127
  const __bf16* ul = up_lo + (size_t)b * NN * (2 * CC);
  const __bf16* ph = ps_hi + (size_t)b * MM * CC;
  const __bf16* pl = ps_lo + (size_t)b * MM * CC;

  // A tiles (16 rows of pcd_up features) are loop-invariant: hoist all K.
  v16bf ah[4], al[4];
#pragma unroll
  for (int ks = 0; ks < 4; ks++) {
    ah[ks] = load_Amat(uh, 2 * CC, n0, ks * 32, lane);
    al[ks] = load_Amat(ul, 2 * CC, n0, ks * 32, lane);
  }
  float q2v[8];
#pragma unroll
  for (int r = 0; r < 8; r++) q2v[r] = q2[(size_t)b * NN + n0 + r + h * 8];

  float d0 = 3.4e38f, d1 = 3.4e38f, d2 = 3.4e38f;
  int i0 = 0x7fffffff, i1 = 0x7fffffff, i2 = 0x7fffffff;

  int srow_t = threadIdx.x >> 4;        // staging row 0..15
  int soff = (threadIdx.x & 15) * 8;    // staging 16B chunk

  for (int m0 = 0; m0 < MM; m0 += 16) {
    // cooperative stage of B panel (all 8 waves share it)
    *(v8bf*)(Bh_s + srow_t * CC + soff) = *(const v8bf*)(ph + (size_t)(m0 + srow_t) * CC + soff);
    *(v8bf*)(Bl_s + srow_t * CC + soff) = *(const v8bf*)(pl + (size_t)(m0 + srow_t) * CC + soff);
    int mn = (m0 + 16) & (MM - 1);
    __builtin_prefetch(ph + (size_t)(mn + srow_t) * CC + soff, 0, 0);
    __builtin_prefetch(pl + (size_t)(mn + srow_t) * CC + soff, 0, 0);
    __syncthreads();

    v8f acc = {};
#pragma unroll
    for (int ks = 0; ks < 4; ks++) {
      v16bf bh = *(const v16bf*)(Bh_s + col * CC + ks * 32 + h * 16);
      v16bf bl = *(const v16bf*)(Bl_s + col * CC + ks * 32 + h * 16);
      acc = wmma3(ah[ks], al[ks], bh, bl, acc);
    }
    float k2v = k2[(size_t)b * MM + m0 + col];
#pragma unroll
    for (int r = 0; r < 8; r++)
      tile[w][r + h * 8][col] = q2v[r] + k2v - 2.f * acc[r];
    __builtin_amdgcn_wave_barrier();  // DS ops in-order per wave; keep scheduler honest
#pragma unroll
    for (int j = 0; j < 8; j++) {
      float d = tile[w][col][cbase + j];
      top3_insert(d, m0 + cbase + j, d0, i0, d1, i1, d2, i2);
    }
    __syncthreads();  // all waves done with B panel before next stage
  }

  // merge the two column halves per row via LDS (intra-wave, DS in-order)
  md[w][lane][0] = d0; md[w][lane][1] = d1; md[w][lane][2] = d2;
  mi[w][lane][0] = i0; mi[w][lane][1] = i1; mi[w][lane][2] = i2;
  __builtin_amdgcn_wave_barrier();
  if (lane < 16) {
#pragma unroll
    for (int t = 0; t < 3; t++)
      top3_insert(md[w][lane + 16][t], mi[w][lane + 16][t], d0, i0, d1, i1, d2, i2);
    float w0 = 1.f / (d0 + 1e-8f), w1 = 1.f / (d1 + 1e-8f), w2 = 1.f / (d2 + 1e-8f);
    float inv = 1.f / (w0 + w1 + w2);
    size_t base = ((size_t)b * NN + n0 + lane) * 3;
    idx_out[base + 0] = i0; idx_out[base + 1] = i1; idx_out[base + 2] = i2;
    w_out[base + 0] = w0 * inv; w_out[base + 1] = w1 * inv; w_out[base + 2] = w2 * inv;
  }
}

// ---------------- K7: weighted 3-NN gather -> x_t cols 128..255 (bf16 hi/lo) ----------
__global__ void k_interp(const float* __restrict__ psc_t, const int* __restrict__ idx,
                         const float* __restrict__ wts, __bf16* __restrict__ xt_hi,
                         __bf16* __restrict__ xt_lo) {
  int c = threadIdx.x;  // 0..127
  int bn = blockIdx.x;  // b*NN + n
  int b = bn >> 13;
  const float* pt = psc_t + (size_t)b * MM * CC;
  size_t base = (size_t)bn * 3;
  int j0 = idx[base], j1 = idx[base + 1], j2 = idx[base + 2];
  float w0 = wts[base], w1 = wts[base + 1], w2 = wts[base + 2];
  float v = w0 * pt[(size_t)j0 * CC + c] + w1 * pt[(size_t)j1 * CC + c] + w2 * pt[(size_t)j2 * CC + c];
  size_t o = (size_t)bn * (2 * CC) + CC + c;
  __bf16 hh = (__bf16)v;
  xt_hi[o] = hh;
  xt_lo[o] = (__bf16)(v - (float)hh);
}

// ---------------- K8: final BN+LeakyReLU in place on d_out ----------------------------
__global__ void k_bn_apply_out(float* __restrict__ out, const float* __restrict__ scale,
                               const float* __restrict__ shift) {
  size_t i = (size_t)blockIdx.x * 256 + threadIdx.x;
  int ch = (int)((i >> 13) & (CC - 1));
  float y = out[i] * scale[ch] + shift[ch];
  out[i] = y > 0.f ? y : 0.2f * y;
}

extern "C" void kernel_launch(void* const* d_in, const int* in_sizes, int n_in,
                              void* d_out, int out_size, void* d_ws, size_t ws_size,
                              hipStream_t stream) {
  const float* pcd_up = (const float*)d_in[0];        // [8,128,8192]
  const float* points_select = (const float*)d_in[1]; // [8,128,2048]
  // d_in[2], d_in[3]: xyz tensors, unused by the reference math
  const float* w_conv = (const float*)d_in[4];        // [128,128]
  const float* g1 = (const float*)d_in[5];
  const float* b1 = (const float*)d_in[6];
  const float* w_res = (const float*)d_in[7];         // [128,256]
  const float* g2 = (const float*)d_in[8];
  const float* b2 = (const float*)d_in[9];
  float* out = (float*)d_out;

  char* p = (char*)d_ws;
  auto alloc = [&](size_t bytes) { void* r = (void*)p; p += (bytes + 255) & ~(size_t)255; return r; };

  __bf16* wc_hi = (__bf16*)alloc((size_t)CC * CC * 2);
  __bf16* wc_lo = (__bf16*)alloc((size_t)CC * CC * 2);
  __bf16* wr_hi = (__bf16*)alloc((size_t)CC * 2 * CC * 2);
  __bf16* wr_lo = (__bf16*)alloc((size_t)CC * 2 * CC * 2);
  __bf16* ps_t_hi = (__bf16*)alloc((size_t)BB * MM * CC * 2);
  __bf16* ps_t_lo = (__bf16*)alloc((size_t)BB * MM * CC * 2);
  __bf16* x_t_hi = (__bf16*)alloc((size_t)BB * NN * 2 * CC * 2);
  __bf16* x_t_lo = (__bf16*)alloc((size_t)BB * NN * 2 * CC * 2);
  float* psc_pre = (float*)alloc((size_t)BB * CC * MM * 4);
  float* psc_t = (float*)alloc((size_t)BB * MM * CC * 4);
  float* q2 = (float*)alloc((size_t)BB * NN * 4);
  float* k2 = (float*)alloc((size_t)BB * MM * 4);
  float* scale1 = (float*)alloc(CC * 4);
  float* shift1 = (float*)alloc(CC * 4);
  float* scale2 = (float*)alloc(CC * 4);
  float* shift2 = (float*)alloc(CC * 4);
  int* knn_idx = (int*)alloc((size_t)BB * NN * 3 * 4);
  float* knn_w = (float*)alloc((size_t)BB * NN * 3 * 4);

  // 0) weights -> bf16 hi/lo
  k_prep_weights<<<192, 256, 0, stream>>>(w_conv, w_res, wc_hi, wc_lo, wr_hi, wr_lo);
  // 1) transpose points_select -> ps_t (rowStride 128)
  k_transpose_split<CC, MM, CC><<<BB * (CC / 32) * (MM / 32), 256, 0, stream>>>(points_select, ps_t_hi, ps_t_lo);
  // 2) transpose pcd_up -> x_t cols 0..127 (rowStride 256)
  k_transpose_split<CC, NN, 2 * CC><<<BB * (CC / 32) * (NN / 32), 256, 0, stream>>>(pcd_up, x_t_hi, x_t_lo);
  // 3) squared norms
  k_colnorm2<CC, NN><<<(BB * NN) / 256, 256, 0, stream>>>(pcd_up, q2);
  k_colnorm2<CC, MM><<<(BB * MM) / 256, 256, 0, stream>>>(points_select, k2);
  // 4) conv1: psc_pre = w_conv @ points_select  (bf16x3 WMMA, LDS-shared B)
  k_conv_wmma<CC, MM, CC, CC><<<BB * (MM / 16), 256, 0, stream>>>(wc_hi, wc_lo, ps_t_hi, ps_t_lo, psc_pre);
  // 5) BN1 stats + apply (+transpose to [b][m][c])
  k_bn_stats<MM><<<CC, 256, 0, stream>>>(psc_pre, g1, b1, scale1, shift1);
  k_bn_apply_psct<<<(BB * MM * CC) / 256, 256, 0, stream>>>(psc_pre, scale1, shift1, psc_t);
  // 6) fused distance GEMM + top-3
  k_dist_topk<<<(BB * (NN / 16)) / 8, 256, 0, stream>>>(x_t_hi, x_t_lo, ps_t_hi, ps_t_lo,
                                                        q2, k2, knn_idx, knn_w);
  // 7) 3-NN interpolation into x_t cols 128..255
  k_interp<<<BB * NN, CC, 0, stream>>>(psc_t, knn_idx, knn_w, x_t_hi, x_t_lo);
  // 8) conv2: out_pre = w_res @ concat(pcd_up, interpolated)  (bf16x3 WMMA, LDS-shared B)
  k_conv_wmma<CC, NN, 2 * CC, 2 * CC><<<BB * (NN / 16), 256, 0, stream>>>(wr_hi, wr_lo, x_t_hi, x_t_lo, out);
  // 9) BN2 stats + apply in place
  k_bn_stats<NN><<<CC, 256, 0, stream>>>(out, g2, b2, scale2, shift2);
  k_bn_apply_out<<<(BB * CC * NN) / 256, 256, 0, stream>>>(out, scale2, shift2);

  (void)in_sizes; (void)n_in; (void)out_size; (void)ws_size;
}
